// LIAFConvCell_34600256537585
// MI455X (gfx1250) — matrix operand order repaired and checked by reading, not compiled, not run on hardware
//
#include <hip/hip_runtime.h>

// LIAF conv cell for MI455X (gfx1250, wave32, WMMA).
// (1) weight repack fp32 -> bf16 hi/lo GEMM-B layout
// (2) implicit-GEMM 3x3 conv via v_wmma_f32_16x16x32_bf16, bf16x3 compensated
//     products (~fp32 accuracy), double-buffered LDS, async global->LDS weight
//     staging (ASYNCcnt), register-prefetched im2col, fused BN statistics
// (3) stats -> per-channel scale/shift
// (4) BN + LIF recurrence over T, in place on d_out

#define CIN   64
#define COUT  128
#define Bb    16
#define Tt    8
#define Hh    56
#define Ww    56
#define HW    3136
#define KCH   576      // CIN * 9
#define NKCH  18       // 576 / 32

typedef __attribute__((ext_vector_type(16))) __bf16 v16bf;
typedef __attribute__((ext_vector_type(8)))  float  v8f;
typedef int v4i_vs __attribute__((__vector_size__(4 * sizeof(int))));

#if defined(__has_builtin)
#  if __has_builtin(__builtin_amdgcn_global_load_async_to_lds_b128) && \
      __has_builtin(__builtin_amdgcn_s_wait_asynccnt)
#    define USE_ASYNC_LDS 1
#  endif
#endif
#ifndef USE_ASYNC_LDS
#  define USE_ASYNC_LDS 0
#endif

#if USE_ASYNC_LDS
// Per-lane async copy of 16B global -> LDS, tracked by ASYNCcnt.
__device__ __forceinline__ void async_cp16(void* lds, const void* g) {
  __builtin_amdgcn_global_load_async_to_lds_b128(
      (__attribute__((address_space(1))) v4i_vs*)g,
      (__attribute__((address_space(3))) v4i_vs*)lds,
      0, 0);
}
#endif

__device__ __forceinline__ unsigned short f2bf(float f) {
  unsigned int u = __float_as_uint(f);
  u += 0x7fffu + ((u >> 16) & 1u);           // round-to-nearest-even
  return (unsigned short)(u >> 16);
}
__device__ __forceinline__ float bf2f(unsigned short h) {
  return __uint_as_float(((unsigned int)h) << 16);
}

// A-fragment (16-bit A 16x32): lanes 0-15 -> dwords {0..3, 8..11},
// lanes 16-31 -> dwords {4..7, 12..15} of the 32-element K row.
__device__ __forceinline__ v16bf load_frag_a(const unsigned short* rowp, int half) {
  union { uint4 q[2]; v16bf v; } u;
  const uint4* p = (const uint4*)rowp;
  u.q[0] = p[half];
  u.q[1] = p[half + 2];
  return u.v;
}
// B-fragment (16-bit B 32x16): lanes 0-15 -> dwords {0..3, 8..11},
// lanes 16-31 -> dwords {4..7, 12..15} (same K order as A).
__device__ __forceinline__ v16bf load_frag_b(const unsigned short* rowp, int half) {
  union { uint4 q[2]; v16bf v; } u;
  const uint4* p = (const uint4*)rowp;
  u.q[0] = p[half];
  u.q[1] = p[half + 2];
  return u.v;
}

// ---------------- K0: weight repack + stats zeroing ----------------
// wt layout: [ks 0..17][n 0..127][kl 0..31], k = (kh*3+kw)*64 + cin.
__global__ void wt_transform(const float* __restrict__ w,
                             unsigned short* __restrict__ wt_hi,
                             unsigned short* __restrict__ wt_lo,
                             float* __restrict__ stats) {
  const int gid = blockIdx.x * 256 + threadIdx.x;
  if (blockIdx.x == 0) stats[threadIdx.x] = 0.0f;   // zero sum[128]+sumsq[128]
  if (gid >= KCH * COUT) return;
  const int kl  = gid & 31;
  const int n   = (gid >> 5) & 127;
  const int ks  = gid >> 12;
  const int tap = ks >> 1;
  const int kh  = tap / 3, kw = tap % 3;
  const int cin = ((ks & 1) << 5) + kl;
  const float f = w[((n * CIN + cin) * 3 + kh) * 3 + kw];
  const unsigned short h = f2bf(f);
  wt_hi[gid] = h;
  wt_lo[gid] = f2bf(f - bf2f(h));
}

// ---------------- K1: implicit-GEMM conv + fused BN stats ----------------
// Block tile: M = 128 (2 image rows x 64, 56 valid each) x N = 128 cout.
// 8 waves = 2(M-row) x 4(N-slice of 32). Per wave: 4 M-subtiles x 2 N-subtiles.
// Software pipeline: async B copy + A register prefetch for chunk ks+1 issued
// before the 24 WMMAs of chunk ks; A convert/store after; one barrier/iter.
__global__ __launch_bounds__(256)
void conv_wmma(const float* __restrict__ xg,
               const unsigned short* __restrict__ wt_hi,
               const unsigned short* __restrict__ wt_lo,
               const float* __restrict__ bias,
               float* __restrict__ yout,
               float* __restrict__ stats) {
  __shared__ unsigned short As_hi[2][128][32];
  __shared__ unsigned short As_lo[2][128][32];
  __shared__ unsigned short Bs_hi[2][128][32];
  __shared__ unsigned short Bs_lo[2][128][32];
  __shared__ float ssum[COUT];
  __shared__ float ssq[COUT];

  const int tid = threadIdx.x;
  int bid = blockIdx.x;
  const int yhalf = bid % 28; bid /= 28;
  const int t = bid % Tt;
  const int b = bid / Tt;
  const int y0 = yhalf * 2;

  if (tid < COUT) { ssum[tid] = 0.0f; ssq[tid] = 0.0f; }

  const int lane = tid & 31;
  const int wid  = tid >> 5;
  const int wm   = wid >> 2;      // which image row of the tile
  const int wn   = wid & 3;       // which 32-cout slice
  const int half = lane >> 4;
  const int lr   = lane & 15;

  const int kl4 = tid >> 6;       // 0..3
  const int xm  = tid & 63;       // x within padded 64-wide row

  float pf[2][8];                 // A prefetch registers (2 rows x 8 cin)

  // Stage the 2x8KB hi/lo weight tile for chunk ks into LDS buffer bufi.
  auto stage_b = [&](int bufi, int ks) {
    const uint4* gh = (const uint4*)(wt_hi + ks * 4096);
    const uint4* gl = (const uint4*)(wt_lo + ks * 4096);
    uint4* sh = (uint4*)&Bs_hi[bufi][0][0];
    uint4* sl = (uint4*)&Bs_lo[bufi][0][0];
#if USE_ASYNC_LDS
    async_cp16(sh + tid,       gh + tid);
    async_cp16(sh + tid + 256, gh + tid + 256);
    async_cp16(sl + tid,       gl + tid);
    async_cp16(sl + tid + 256, gl + tid + 256);
#else
    sh[tid]       = gh[tid];
    sh[tid + 256] = gh[tid + 256];
    sl[tid]       = gl[tid];
    sl[tid + 256] = gl[tid + 256];
#endif
  };

  // Issue the 16 W-contiguous global loads of the im2col tile for chunk ks.
  auto load_a_regs = [&](int ks) {
    const int tap = ks >> 1;
    const int kh  = tap / 3, kw = tap % 3;
    const int cb  = (ks & 1) << 5;
#pragma unroll
    for (int r = 0; r < 2; ++r) {
      const int y_im = y0 + r + kh - 1;
      const int x_im = xm + kw - 1;
      const bool valid = (xm < Ww) && (y_im >= 0) && (y_im < Hh) &&
                         (x_im >= 0) && (x_im < Ww);
      const float* src = xg + (((size_t)(b * CIN + cb + kl4) * Tt + t) * HW) +
                         y_im * Ww + x_im;
#pragma unroll
      for (int i = 0; i < 8; ++i)
        pf[r][i] = valid ? src[(size_t)i * 4 * Tt * HW] : 0.0f;
    }
  };

  // bf16 hi/lo split of the prefetched tile into LDS buffer bufi.
  auto store_a_lds = [&](int bufi) {
#pragma unroll
    for (int r = 0; r < 2; ++r) {
      const int m = r * 64 + xm;
#pragma unroll
      for (int i = 0; i < 8; ++i) {
        const int kl = i * 4 + kl4;
        const unsigned short h = f2bf(pf[r][i]);
        As_hi[bufi][m][kl] = h;
        As_lo[bufi][m][kl] = f2bf(pf[r][i] - bf2f(h));
      }
    }
  };

  v8f acc[4][2];
#pragma unroll
  for (int i = 0; i < 4; ++i)
#pragma unroll
    for (int j = 0; j < 2; ++j)
#pragma unroll
      for (int k = 0; k < 8; ++k) acc[i][j][k] = 0.0f;

  // ---- prologue: stage chunk 0 ----
  stage_b(0, 0);
  load_a_regs(0);
  store_a_lds(0);
#if USE_ASYNC_LDS
  __builtin_amdgcn_s_wait_asynccnt(0);
#endif
  __syncthreads();

  for (int ks = 0; ks < NKCH; ++ks) {
    const int cur = ks & 1;
    const int nxt = cur ^ 1;
    const bool pre = (ks + 1) < NKCH;

    if (pre) {
      stage_b(nxt, ks + 1);      // async copy overlaps WMMAs below
      load_a_regs(ks + 1);       // global loads in flight during WMMAs
    }

    // ---- fragments + 24 WMMAs (bf16x3 products, f32 accumulate) ----
    v16bf ah[4], al[4], bh[2], bl[2];
#pragma unroll
    for (int i = 0; i < 4; ++i) {
      const int row = wm * 64 + i * 16 + lr;
      ah[i] = load_frag_a(&As_hi[cur][row][0], half);
      al[i] = load_frag_a(&As_lo[cur][row][0], half);
    }
#pragma unroll
    for (int j = 0; j < 2; ++j) {
      const int row = wn * 32 + j * 16 + lr;
      bh[j] = load_frag_b(&Bs_hi[cur][row][0], half);
      bl[j] = load_frag_b(&Bs_lo[cur][row][0], half);
    }
#pragma unroll
    for (int i = 0; i < 4; ++i)
#pragma unroll
      for (int j = 0; j < 2; ++j) {
        acc[i][j] = __builtin_amdgcn_wmma_f32_16x16x32_bf16(
            false, ah[i], false, bh[j], (short)0, acc[i][j], false, false);
        acc[i][j] = __builtin_amdgcn_wmma_f32_16x16x32_bf16(
            false, al[i], false, bh[j], (short)0, acc[i][j], false, false);
        acc[i][j] = __builtin_amdgcn_wmma_f32_16x16x32_bf16(
            false, ah[i], false, bl[j], (short)0, acc[i][j], false, false);
      }

    if (pre) store_a_lds(nxt);   // VALU split co-executes after WMMA issue
#if USE_ASYNC_LDS
    __builtin_amdgcn_s_wait_asynccnt(0);
#endif
    __syncthreads();
  }

  // ---- epilogue: bias, store conv output, fused BN partial sums ----
  const int y = y0 + wm;
#pragma unroll
  for (int j = 0; j < 2; ++j) {
    const int c = wn * 32 + j * 16 + lr;          // D layout: N = lane&15
    const float bia = bias[c];
    float psum = 0.0f, psq = 0.0f;
    float* obase = yout + (((size_t)(b * COUT + c) * Tt + t) * HW) + y * Ww;
#pragma unroll
    for (int i = 0; i < 4; ++i) {
      const int xb = i * 16 + half * 8;           // D layout: M = vgpr (+8 hi half)
      if (xb >= Ww) continue;                     // padded columns
      float v[8];
#pragma unroll
      for (int k = 0; k < 8; ++k) {
        const float f = acc[i][j][k] + bia;
        v[k] = f;
        psum += f;
        psq  += f * f;
      }
      float4* o = (float4*)(obase + xb);
      o[0] = make_float4(v[0], v[1], v[2], v[3]);
      o[1] = make_float4(v[4], v[5], v[6], v[7]);
    }
    atomicAdd(&ssum[c], psum);
    atomicAdd(&ssq[c],  psq);
  }
  __syncthreads();
  if (tid < COUT) {
    atomicAdd(&stats[tid],        ssum[tid]);
    atomicAdd(&stats[COUT + tid], ssq[tid]);
  }
}

// ---------------- K2: stats -> per-channel scale/shift ----------------
__global__ void bn_stats(const float* __restrict__ stats,
                         const float* __restrict__ gamma,
                         const float* __restrict__ beta,
                         float* __restrict__ scaleshift) {
  const int c = threadIdx.x;
  if (c >= COUT) return;
  const float Ninv = 1.0f / (float)(Bb * Tt * HW);
  const float mean = stats[c] * Ninv;
  const float var  = stats[COUT + c] * Ninv - mean * mean;
  const float inv  = rsqrtf(var + 1e-5f);
  const float sc   = gamma[c] * inv;
  scaleshift[c]        = sc;
  scaleshift[COUT + c] = beta[c] - mean * sc;
}

// ---------------- K3: BN + LIF recurrence, in place on d_out ----------------
__global__ __launch_bounds__(256)
void bn_lif(float* __restrict__ y,
            const float* __restrict__ scaleshift,
            const float* __restrict__ decay) {
  const long long gid = (long long)blockIdx.x * 256 + threadIdx.x;
  const long long total = (long long)Bb * COUT * HW;
  if (gid >= total) return;
  const int s  = (int)(gid % HW);
  const int c  = (int)((gid / HW) % COUT);
  const int bb = (int)(gid / ((long long)HW * COUT));
  const float sc   = scaleshift[c];
  const float sh   = scaleshift[COUT + c];
  const float sigd = 1.0f / (1.0f + __expf(-decay[0]));
  float* p = y + (((size_t)(bb * COUT + c) * Tt) * HW) + s;
  float mem = 0.0f, spike = 0.0f;
#pragma unroll
  for (int t = 0; t < Tt; ++t) {
    const float v = p[(size_t)t * HW] * sc + sh;
    mem = mem * sigd * (1.0f - spike) + v;
    spike = (mem > 0.5f) ? 1.0f : 0.0f;
    p[(size_t)t * HW] = fmaxf(mem, 0.0f);
  }
}

extern "C" void kernel_launch(void* const* d_in, const int* in_sizes, int n_in,
                              void* d_out, int out_size, void* d_ws, size_t ws_size,
                              hipStream_t stream) {
  const float* x      = (const float*)d_in[0];
  const float* conv_w = (const float*)d_in[1];
  const float* conv_b = (const float*)d_in[2];
  const float* gamma  = (const float*)d_in[3];
  const float* beta   = (const float*)d_in[4];
  const float* decay  = (const float*)d_in[5];
  float* out = (float*)d_out;

  // workspace layout (16B aligned): wt_hi | wt_lo | stats[256] | scaleshift[256]
  unsigned short* wt_hi = (unsigned short*)d_ws;
  unsigned short* wt_lo = wt_hi + (size_t)KCH * COUT;
  float* stats      = (float*)(wt_lo + (size_t)KCH * COUT);
  float* scaleshift = stats + 256;

  wt_transform<<<(KCH * COUT) / 256, 256, 0, stream>>>(conv_w, wt_hi, wt_lo, stats);
  conv_wmma<<<Bb * Tt * (Hh / 2), 256, 0, stream>>>(x, wt_hi, wt_lo, conv_b, out, stats);
  bn_stats<<<1, 128, 0, stream>>>(stats, gamma, beta, scaleshift);
  const long long total = (long long)Bb * COUT * HW;
  bn_lif<<<(int)((total + 255) / 256), 256, 0, stream>>>(out, scaleshift, decay);
}
